// Attention_22814866276429
// MI455X (gfx1250) — matrix operand order repaired and checked by reading, not compiled, not run on hardware
//
#include <hip/hip_runtime.h>
#include <hip/hip_bf16.h>
#include <stdint.h>

typedef __attribute__((ext_vector_type(16))) _Float16 v16h;
typedef __attribute__((ext_vector_type(8)))  _Float16 v8h;
typedef __attribute__((ext_vector_type(8)))  float    v8f;
typedef __attribute__((ext_vector_type(4)))  int      v4i;

#define BB 4
#define SS 2048
#define DIM 2048
#define NH 16
#define NKV 4
#define HD 128
#define QKV_N 3072

// ---------------------------------------------------------------------------
// CDNA5 async global->LDS copy (ASYNCcnt-tracked). 16 bytes per lane.
// ---------------------------------------------------------------------------
#if __has_builtin(__builtin_amdgcn_global_load_async_to_lds_b128)
#define ASYNC_VIA_BUILTIN 1
#else
#define ASYNC_VIA_BUILTIN 0
#endif

__device__ __forceinline__ void async_copy16(const _Float16* g, _Float16* l)
{
#if ASYNC_VIA_BUILTIN
    __builtin_amdgcn_global_load_async_to_lds_b128(
        (__attribute__((address_space(1))) v4i*)(unsigned long long)(const void*)g,
        (__attribute__((address_space(3))) v4i*)
            (__attribute__((address_space(3))) _Float16*)l,
        0, 0);
#else
    unsigned lo = (unsigned)(unsigned long long)(__attribute__((address_space(3))) _Float16*)l;
    asm volatile("global_load_async_to_lds_b128 %0, %1, off"
                 :: "v"(lo), "v"((unsigned long long)(const void*)g) : "memory");
#endif
}

#if __has_builtin(__builtin_amdgcn_s_wait_asynccnt)
#define WAIT_ASYNCCNT(n) __builtin_amdgcn_s_wait_asynccnt(n)
#else
#define WAIT_ASYNCCNT(n) asm volatile("s_wait_asynccnt %0" :: "n"(n) : "memory")
#endif

// ---------------------------------------------------------------------------
// f32 -> f16 pre-conversion (one pass over x, wqkv, wo)
// ---------------------------------------------------------------------------
__global__ __launch_bounds__(256) void cvt_f32_f16(
    const float* __restrict__ in, _Float16* __restrict__ out)
{
    size_t i = ((size_t)blockIdx.x * 256 + threadIdx.x) * 8;
    float4 a = *(const float4*)(in + i);
    float4 b = *(const float4*)(in + i + 4);
    v8h o;
    o[0] = (_Float16)a.x; o[1] = (_Float16)a.y; o[2] = (_Float16)a.z; o[3] = (_Float16)a.w;
    o[4] = (_Float16)b.x; o[5] = (_Float16)b.y; o[6] = (_Float16)b.z; o[7] = (_Float16)b.w;
    *(v8h*)(out + i) = o;
}

// ---------------------------------------------------------------------------
// GEMM: C[M,N](f32) = A[M,K](f16) * B[N,K](f16)^T
// Block tile 128x128, 8 waves (4 M x 2 N), wave tile 32x64 (2x4 WMMA tiles),
// async double-buffered LDS staging.  8 WMMAs per wave per K-step.
// ---------------------------------------------------------------------------
__global__ __launch_bounds__(256) void gemm_bt_wmma_f16(
    const _Float16* __restrict__ A, const _Float16* __restrict__ B,
    float* __restrict__ C, int M, int N, int K)
{
    __shared__ _Float16 As[2][128][32];   // 16 KB
    __shared__ _Float16 Bs[2][128][32];   // 16 KB

    const int tid  = threadIdx.x;
    const int wave = tid >> 5, lane = tid & 31;
    const int wm = wave >> 1, wn = wave & 1;     // 4 x 2 wave grid
    const int hx = lane >> 4, l16 = lane & 15;
    const int bm = blockIdx.y * 128, bn = blockIdx.x * 128;

    // staging map: 16B chunks; each tile is 128 rows x 4 chunks = 512 chunks,
    // 256 threads -> 2 chunks per matrix per thread.
    const int r0 = tid >> 2, cc = (tid & 3) * 8;
    const _Float16* gA0 = A + (size_t)(bm + r0) * K + cc;
    const _Float16* gA1 = A + (size_t)(bm + r0 + 64) * K + cc;
    const _Float16* gB0 = B + (size_t)(bn + r0) * K + cc;
    const _Float16* gB1 = B + (size_t)(bn + r0 + 64) * K + cc;

    auto stage = [&](int buf, int k0) {
        async_copy16(gA0 + k0, &As[buf][r0][cc]);
        async_copy16(gA1 + k0, &As[buf][r0 + 64][cc]);
        async_copy16(gB0 + k0, &Bs[buf][r0][cc]);
        async_copy16(gB1 + k0, &Bs[buf][r0 + 64][cc]);
    };

    const int nk = K / 32;
    stage(0, 0);

    v8f acc[2][4] = {};
    for (int t = 0; t < nk; t++) {
        const int buf = t & 1;
        if (t + 1 < nk) {
            stage(buf ^ 1, (t + 1) * 32);
            WAIT_ASYNCCNT(4);   // my 4 newest (next tile) pending; current tile done
        } else {
            WAIT_ASYNCCNT(0);
        }
        __syncthreads();        // all waves' current-tile asyncs complete

        v16h af[2], bf[4];
        #pragma unroll
        for (int ms = 0; ms < 2; ms++) {
            int row = wm * 32 + ms * 16 + l16;
            #pragma unroll
            for (int v = 0; v < 8; v++) {
                int kk = ((v < 4) ? 0 : 16) + hx * 8 + (v & 3) * 2;
                af[ms][2 * v]     = As[buf][row][kk];
                af[ms][2 * v + 1] = As[buf][row][kk + 1];
            }
        }
        #pragma unroll
        for (int ns = 0; ns < 4; ns++) {
            int col = wn * 64 + ns * 16 + l16;
            #pragma unroll
            for (int v = 0; v < 8; v++) {
                int kk = ((v < 4) ? 0 : 16) + hx * 8 + (v & 3) * 2;
                bf[ns][2 * v]     = Bs[buf][col][kk];
                bf[ns][2 * v + 1] = Bs[buf][col][kk + 1];
            }
        }
        #pragma unroll
        for (int ms = 0; ms < 2; ms++)
            #pragma unroll
            for (int ns = 0; ns < 4; ns++)
                acc[ms][ns] = __builtin_amdgcn_wmma_f32_16x16x32_f16(
                    false, af[ms], false, bf[ns], (short)0, acc[ms][ns], false, false);
        __syncthreads();        // compute done before next-iter stage overwrites buf
    }

    #pragma unroll
    for (int ms = 0; ms < 2; ms++)
        #pragma unroll
        for (int ns = 0; ns < 4; ns++)
            #pragma unroll
            for (int v = 0; v < 8; v++) {
                int row = bm + wm * 32 + ms * 16 + v + 8 * hx;
                int col = bn + wn * 64 + ns * 16 + l16;
                C[(size_t)row * N + col] = acc[ms][ns][v];
            }
}

// ---------------------------------------------------------------------------
// RMSNorm (q,k) + RoPE + pack q/k/v head-major f16. One wave per head-vector.
// ---------------------------------------------------------------------------
__global__ __launch_bounds__(256) void norm_rope_pack(
    const float* __restrict__ qkv, const float* __restrict__ qw,
    const float* __restrict__ kw, const float* __restrict__ fcos,
    const float* __restrict__ fsin,
    _Float16* __restrict__ Qh, _Float16* __restrict__ Kh, _Float16* __restrict__ Vh)
{
    const int wg   = blockIdx.x * 8 + (threadIdx.x >> 5);
    const int lane = threadIdx.x & 31;
    const int t = wg / 24;
    const int h = wg % 24;
    const int b = t >> 11;
    const int s = t & (SS - 1);
    const int d0 = lane * 4;

    const float* src;
    if (h < 16)      src = qkv + (size_t)t * QKV_N + h * HD;
    else if (h < 20) src = qkv + (size_t)t * QKV_N + 2048 + (h - 16) * HD;
    else             src = qkv + (size_t)t * QKV_N + 2560 + (h - 20) * HD;

    float x[4];
    #pragma unroll
    for (int i = 0; i < 4; i++) x[i] = src[d0 + i];

    if (h < 20) {
        float ssum = x[0]*x[0] + x[1]*x[1] + x[2]*x[2] + x[3]*x[3];
        #pragma unroll
        for (int off = 16; off >= 1; off >>= 1) ssum += __shfl_xor(ssum, off, 32);
        float r = rsqrtf(ssum * (1.0f / 128.0f) + 1.1920928955078125e-07f);
        const float* w = (h < 16) ? qw : kw;
        #pragma unroll
        for (int i = 0; i < 4; i++) x[i] = x[i] * r * w[d0 + i];
        const float* cs = fcos + (size_t)s * HD + d0;
        const float* sn = fsin + (size_t)s * HD + d0;
        float y0 = x[0] * cs[0] - x[1] * sn[0];
        float y1 = x[1] * cs[1] + x[0] * sn[1];
        float y2 = x[2] * cs[2] - x[3] * sn[2];
        float y3 = x[3] * cs[3] + x[2] * sn[3];
        x[0] = y0; x[1] = y1; x[2] = y2; x[3] = y3;
    }

    _Float16* dst;
    if (h < 16)      dst = Qh + (((size_t)b * NH  + h)        * SS + s) * HD;
    else if (h < 20) dst = Kh + (((size_t)b * NKV + (h - 16)) * SS + s) * HD;
    else             dst = Vh + (((size_t)b * NKV + (h - 20)) * SS + s) * HD;
    #pragma unroll
    for (int i = 0; i < 4; i++) dst[d0 + i] = (_Float16)x[i];
}

// ---------------------------------------------------------------------------
// Flash attention: one wave per (b, head, 16-query tile). GQA kvh = h/4.
// K/V tiles async double-buffered in LDS; 16 WMMAs per 32-key block.
// ---------------------------------------------------------------------------
__global__ __launch_bounds__(32) void flash_attn(
    const _Float16* __restrict__ Qh, const _Float16* __restrict__ Kh,
    const _Float16* __restrict__ Vh, const unsigned char* __restrict__ vis,
    _Float16* __restrict__ O /* [B,S, H*HD] */)
{
    __shared__ _Float16 Kt[2][32][HD];  // 16 KB
    __shared__ _Float16 Vt[2][32][HD];  // 16 KB
    __shared__ _Float16 Pt[16][32];     //  1 KB

    const int lane = threadIdx.x;
    const int hx = lane >> 4, l16 = lane & 15;
    const int q0  = blockIdx.x * 16;
    const int hq  = blockIdx.y;
    const int b   = blockIdx.z;
    const int kvh = hq >> 2;

    const _Float16* Qb = Qh + ((size_t)b * NH  + hq)  * SS * HD;
    const _Float16* Kb = Kh + ((size_t)b * NKV + kvh) * SS * HD;
    const _Float16* Vb = Vh + ((size_t)b * NKV + kvh) * SS * HD;
    const unsigned char* visb = vis + (size_t)b * SS;

    auto stageKV = [&](int buf, int kb) {
        const _Float16* kr = Kb + (size_t)(kb + lane) * HD;
        const _Float16* vr = Vb + (size_t)(kb + lane) * HD;
        _Float16* kd = &Kt[buf][lane][0];
        _Float16* vd = &Vt[buf][lane][0];
        #pragma unroll
        for (int i = 0; i < 16; i++) {          // 16 x 16B = one 256B row
            async_copy16(kr + i * 8, kd + i * 8);
            async_copy16(vr + i * 8, vd + i * 8);
        }
    };

    // Q fragments: 4 k-chunks of 32 over head_dim
    v16h qf[4];
    {
        const _Float16* qr = Qb + (size_t)(q0 + l16) * HD;
        #pragma unroll
        for (int kc = 0; kc < 4; kc++)
            #pragma unroll
            for (int v = 0; v < 8; v++) {
                int kk = kc * 32 + ((v < 4) ? 0 : 16) + hx * 8 + (v & 3) * 2;
                qf[kc][2 * v]     = qr[kk];
                qf[kc][2 * v + 1] = qr[kk + 1];
            }
    }

    v8f o[8] = {};
    float m[8], l[8];
    #pragma unroll
    for (int v = 0; v < 8; v++) { m[v] = -1e30f; l[v] = 0.0f; }
    const float scale = 0.08838834764831845f;  // 1/sqrt(128)

    stageKV(0, 0);
    const int NT = SS / 32;
    for (int t = 0; t < NT; t++) {
        const int buf = t & 1;
        const int kb  = t * 32;
        if (t + 1 < NT) {
            stageKV(buf ^ 1, kb + 32);
            WAIT_ASYNCCNT(32);    // next tile (32 ops) pending; current tile landed
        } else {
            WAIT_ASYNCCNT(0);
        }

        // scores: S = Q * K^T
        v8f sc[2] = {};
        #pragma unroll
        for (int ns = 0; ns < 2; ns++) {
            int key = ns * 16 + l16;
            #pragma unroll
            for (int kc = 0; kc < 4; kc++) {
                v16h bf;
                #pragma unroll
                for (int v = 0; v < 8; v++) {
                    int dd = kc * 32 + ((v < 4) ? 0 : 16) + hx * 8 + (v & 3) * 2;
                    bf[2 * v]     = Kt[buf][key][dd];
                    bf[2 * v + 1] = Kt[buf][key][dd + 1];
                }
                sc[ns] = __builtin_amdgcn_wmma_f32_16x16x32_f16(
                    false, qf[kc], false, bf, (short)0, sc[ns], false, false);
            }
        }

        // scale + key-visibility mask
        #pragma unroll
        for (int ns = 0; ns < 2; ns++) {
            bool kvis = visb[kb + ns * 16 + l16] != 0;
            #pragma unroll
            for (int v = 0; v < 8; v++)
                sc[ns][v] = kvis ? sc[ns][v] * scale : -1e30f;
        }

        // online softmax (row = v + 8*hx, spread over the 16 lanes of the half)
        #pragma unroll
        for (int v = 0; v < 8; v++) {
            float mv = fmaxf(sc[0][v], sc[1][v]);
            #pragma unroll
            for (int off = 8; off >= 1; off >>= 1) mv = fmaxf(mv, __shfl_xor(mv, off, 32));
            float mnew = fmaxf(m[v], mv);
            float corr = __expf(m[v] - mnew);
            m[v] = mnew;
            float p0 = __expf(sc[0][v] - mnew);
            float p1 = __expf(sc[1][v] - mnew);
            sc[0][v] = p0; sc[1][v] = p1;
            float rs = p0 + p1;
            #pragma unroll
            for (int off = 8; off >= 1; off >>= 1) rs += __shfl_xor(rs, off, 32);
            l[v] = l[v] * corr + rs;
            #pragma unroll
            for (int nt = 0; nt < 8; nt++) o[nt][v] *= corr;
        }

        // P: C-layout -> A-fragment layout via LDS (in-order DS within wave)
        #pragma unroll
        for (int ns = 0; ns < 2; ns++)
            #pragma unroll
            for (int v = 0; v < 8; v++)
                Pt[v + 8 * hx][ns * 16 + l16] = (_Float16)sc[ns][v];

        v16h pf;
        #pragma unroll
        for (int v = 0; v < 8; v++) {
            int kk = ((v < 4) ? 0 : 16) + hx * 8 + (v & 3) * 2;
            pf[2 * v]     = Pt[l16][kk];
            pf[2 * v + 1] = Pt[l16][kk + 1];
        }

        // O += P * V
        #pragma unroll
        for (int nt = 0; nt < 8; nt++) {
            v16h vf;
            #pragma unroll
            for (int v = 0; v < 8; v++) {
                int kk = ((v < 4) ? 0 : 16) + hx * 8 + (v & 3) * 2;
                vf[2 * v]     = Vt[buf][kk][nt * 16 + l16];
                vf[2 * v + 1] = Vt[buf][kk + 1][nt * 16 + l16];
            }
            o[nt] = __builtin_amdgcn_wmma_f32_16x16x32_f16(
                false, pf, false, vf, (short)0, o[nt], false, false);
        }
    }

    // epilogue: normalize, zero invisible queries, pack f16 [B,S,H*HD]
    #pragma unroll
    for (int v = 0; v < 8; v++) {
        int s = q0 + v + 8 * hx;
        float f = (visb[s] != 0 && l[v] > 0.0f) ? (1.0f / l[v]) : 0.0f;
        #pragma unroll
        for (int nt = 0; nt < 8; nt++) {
            int d = nt * 16 + l16;
            O[(((size_t)b * SS + s) * DIM) + hq * HD + d] = (_Float16)(o[nt][v] * f);
        }
    }
}

// ---------------------------------------------------------------------------
extern "C" void kernel_launch(void* const* d_in, const int* in_sizes, int n_in,
                              void* d_out, int out_size, void* d_ws, size_t ws_size,
                              hipStream_t stream)
{
    const float* x     = (const float*)d_in[0];   // [B,S,DIM]
    const float* wqkv  = (const float*)d_in[1];   // [3072, 2048]
    const float* wo    = (const float*)d_in[2];   // [2048, 2048]
    const float* qnw   = (const float*)d_in[3];
    const float* knw   = (const float*)d_in[4];
    const float* fcos  = (const float*)d_in[5];
    const float* fsin  = (const float*)d_in[6];
    const unsigned char* vis = (const unsigned char*)d_in[7];  // [B,S] bool
    float* out = (float*)d_out;

    const int M = BB * SS;  // 8192
    char* ws = (char*)d_ws;
    _Float16* Xh    = (_Float16*)ws;  ws += (size_t)M * DIM * 2;               // 32 MB
    _Float16* Wqkvh = (_Float16*)ws;  ws += (size_t)QKV_N * DIM * 2;           // 12 MB
    _Float16* Woh   = (_Float16*)ws;  ws += (size_t)DIM * DIM * 2;             //  8 MB
    float*    qkv   = (float*)ws;     ws += (size_t)M * QKV_N * 4;             // 96 MB
    _Float16* Qh    = (_Float16*)ws;  ws += (size_t)BB * NH  * SS * HD * 2;    // 32 MB
    _Float16* Kh    = (_Float16*)ws;  ws += (size_t)BB * NKV * SS * HD * 2;    //  8 MB
    _Float16* Vh    = (_Float16*)ws;  ws += (size_t)BB * NKV * SS * HD * 2;    //  8 MB
    _Float16* Ao    = (_Float16*)ws;                                           // 32 MB

    // 0) f32 -> f16 pre-conversion (activations + weights)
    cvt_f32_f16<<<(M * DIM) / 2048, 256, 0, stream>>>(x, Xh);
    cvt_f32_f16<<<(QKV_N * DIM) / 2048, 256, 0, stream>>>(wqkv, Wqkvh);
    cvt_f32_f16<<<(DIM * DIM) / 2048, 256, 0, stream>>>(wo, Woh);

    // 1) qkv = x @ wqkv^T
    gemm_bt_wmma_f16<<<dim3(QKV_N / 128, M / 128), 256, 0, stream>>>(
        Xh, Wqkvh, qkv, M, QKV_N, DIM);

    // 2) RMSNorm + RoPE + pack head-major f16
    norm_rope_pack<<<(M * 24) / 8, 256, 0, stream>>>(
        qkv, qnw, knw, fcos, fsin, Qh, Kh, Vh);

    // 3) GQA flash attention
    flash_attn<<<dim3(SS / 16, NH, BB), 32, 0, stream>>>(Qh, Kh, Vh, vis, Ao);

    // 4) out = attn @ wo^T
    gemm_bt_wmma_f16<<<dim3(DIM / 128, M / 128), 256, 0, stream>>>(
        Ao, Woh, out, M, DIM, DIM);
}
